// PromptMaskCrossAttention_38594576122201
// MI455X (gfx1250) — matrix operand order repaired and checked by reading, not compiled, not run on hardware
//
#include <hip/hip_runtime.h>
#include <hip/hip_bf16.h>

// ---------------------------------------------------------------------------
// PromptMaskCrossAttention for gfx1250 (MI455X).
// B=8, N=8, C=256, S=64*64=4096, heads=8, dh=32.
// Heavy path: fused layernorm + K/V projection (bf16 WMMA 16x16x32, f32 acc)
// + single-query attention with online softmax, chunked over S.
// ---------------------------------------------------------------------------

#define CC        256
#define SC        64          // spatial chunk per block
#define NCHUNK    64          // 4096 / 64
#define RA_STRIDE 260         // padded f32 tile row (bank-conflict avoidance)
#define XB_STRIDE 264         // padded bf16 tile row
#define SMEM_BYTES 105984

typedef __attribute__((ext_vector_type(16))) __bf16 v16bf;
typedef __attribute__((ext_vector_type(8)))  float  v8f;

__device__ __forceinline__ __bf16 f2bf(float f) {
  unsigned u = __builtin_bit_cast(unsigned, f);
  u += 0x7FFFu + ((u >> 16) & 1u);            // round-to-nearest-even
  unsigned short h = (unsigned short)(u >> 16);
  return __builtin_bit_cast(__bf16, h);
}

// ---------------------------------------------------------------------------
// Kernel 0: convert Wk, Wv (f32 [256,256] each) to bf16 in workspace.
// ---------------------------------------------------------------------------
__global__ void prep_weights_kernel(const float* __restrict__ Wk,
                                    const float* __restrict__ Wv,
                                    __bf16* __restrict__ bfW) {
  int i = blockIdx.x * 256 + threadIdx.x;     // 0 .. 131071
  if (i < 65536) bfW[i] = f2bf(Wk[i]);
  else           bfW[i] = f2bf(Wv[i - 65536]);
}

// ---------------------------------------------------------------------------
// Kernel 1: layernorm(mask_tokens) and q = normed @ Wq.T + bq.  64 rows only.
// ---------------------------------------------------------------------------
__global__ void stage_q_kernel(const float* __restrict__ mask_tokens,
                               const float* __restrict__ ln_t_w,
                               const float* __restrict__ ln_t_b,
                               const float* __restrict__ Wq,
                               const float* __restrict__ bq,
                               float* __restrict__ q_out) {
  int bn = blockIdx.x;                        // 0..63
  int t  = threadIdx.x;                       // 0..255
  __shared__ float xn[CC];
  __shared__ float red[CC];
  float v = mask_tokens[bn * CC + t];
  red[t] = v; __syncthreads();
  for (int off = 128; off > 0; off >>= 1) {
    if (t < off) red[t] += red[t + off];
    __syncthreads();
  }
  float mu = red[0] * (1.0f / CC);
  __syncthreads();
  float d = v - mu;
  red[t] = d * d; __syncthreads();
  for (int off = 128; off > 0; off >>= 1) {
    if (t < off) red[t] += red[t + off];
    __syncthreads();
  }
  float rstd = rsqrtf(red[0] * (1.0f / CC) + 1e-5f);
  xn[t] = d * rstd * ln_t_w[t] + ln_t_b[t];
  __syncthreads();
  float acc = bq[t];
  const float* wr = Wq + (size_t)t * CC;
  for (int ci = 0; ci < CC; ++ci) acc += xn[ci] * wr[ci];
  q_out[bn * CC + t] = acc;
}

// ---------------------------------------------------------------------------
// bf16 WMMA projection of the normalized tile:  out[64,256] = X @ W.T + bias
// Wave `wave` owns co in [wave*64, wave*64+64).  A/B fragments use contiguous k = 16*half + s.
// ---------------------------------------------------------------------------
__device__ __forceinline__ void project_wmma(const __bf16* __restrict__ xbf,
                                             const __bf16* __restrict__ Wb,
                                             const float*  __restrict__ bias,
                                             float* __restrict__ outT,
                                             int lane, int wave) {
  const int half = lane >> 4;
  const int nrow = lane & 15;
  for (int ct = 0; ct < 4; ++ct) {
    const int co0 = wave * 64 + ct * 16;
    v8f acc[4] = {};
    for (int kt = 0; kt < 8; ++kt) {
      const int ci0 = kt * 32;
      // B fragment: B[ci,co] = W[co,ci]; lane n = nrow, K = 2r + 16*half + {0,1}
      v16bf bf;
      const __bf16* wr = Wb + (size_t)(co0 + nrow) * CC + ci0 + 16 * half;
#pragma unroll
      for (int r = 0; r < 8; ++r) {
        bf[2 * r]     = wr[2 * r];
        bf[2 * r + 1] = wr[2 * r + 1];
      }
#pragma unroll
      for (int mt = 0; mt < 4; ++mt) {
        const __bf16* xr = xbf + (size_t)(mt * 16 + nrow) * XB_STRIDE + ci0;
        v16bf af;
#pragma unroll
        for (int r = 0; r < 8; ++r) {
          const int k = 2 * r + 16 * half;
          af[2 * r]     = xr[k];
          af[2 * r + 1] = xr[k + 1];
        }
        acc[mt] = __builtin_amdgcn_wmma_f32_16x16x32_bf16(
            false, af, false, bf, (short)0, acc[mt], false, false);
      }
    }
    const float bsum = bias[co0 + nrow];
#pragma unroll
    for (int mt = 0; mt < 4; ++mt)
#pragma unroll
      for (int v = 0; v < 8; ++v)
        outT[(size_t)(mt * 16 + v + 8 * half) * RA_STRIDE + co0 + nrow] =
            acc[mt][v] + bsum;
  }
}

// ---------------------------------------------------------------------------
// Kernel 2: per (bn, chunk) fused layernorm + K/V proj + partial softmax ctx.
// grid (chunk=64, bn=64), block 128 (4 waves).
// partial record per (bn,chunk): [m(8) | l(8) | acc(256)] = 272 floats.
// ---------------------------------------------------------------------------
__global__ void __launch_bounds__(128)
attn_partial_kernel(const float* __restrict__ prompt,
                    const float* __restrict__ ln_p_w,
                    const float* __restrict__ ln_p_b,
                    const __bf16* __restrict__ bfWk,
                    const float* __restrict__ bk,
                    const __bf16* __restrict__ bfWv,
                    const float* __restrict__ bv,
                    const float* __restrict__ q_ws,
                    float* __restrict__ part_ws) {
  extern __shared__ __align__(16) char smem_raw[];
  float*  regA   = (float*)(smem_raw);                 // [64][260] f32 X / K / V
  __bf16* xbf    = (__bf16*)(smem_raw + 66560);        // [64][264] normalized bf16
  float*  scores = (float*)(smem_raw + 100352);        // [64][8] (also phase-1 scratch)
  float*  qsm    = (float*)(smem_raw + 102400);        // [256]
  float*  stat   = (float*)(smem_raw + 103424);        // [64][2] mean, rstd
  float*  lnw    = (float*)(smem_raw + 103936);        // [256]
  float*  lnb    = (float*)(smem_raw + 104960);        // [256]

  const int chunk = blockIdx.x;                        // 0..63
  const int bn    = blockIdx.y;                        // 0..63
  const int b = bn >> 3, n = bn & 7;
  const int s0 = chunk * SC;
  const int t = threadIdx.x;                           // 0..127
  const int lane = t & 31;
  const int wave = t >> 5;

  qsm[t]       = q_ws[bn * CC + t];
  qsm[t + 128] = q_ws[bn * CC + t + 128];
  lnw[t] = ln_p_w[t];       lnw[t + 128] = ln_p_w[t + 128];
  lnb[t] = ln_p_b[t];       lnb[t + 128] = ln_p_b[t + 128];

  // ---- Phase 1: coalesced load of the [64 x 256] chunk + row stats --------
  const int srow = t & 63;                             // spatial row in chunk
  const int cpar = t >> 6;                             // channel parity (0/1)
  {
    const float* base = prompt + (size_t)(n * 8 + b) * CC * 4096 + (s0 + srow);
    float sum = 0.f, sumsq = 0.f;
    for (int c = cpar; c < CC; c += 2) {
      float v = base[(size_t)c * 4096];                // lanes span consecutive s
      regA[(size_t)srow * RA_STRIDE + c] = v;
      sum += v; sumsq += v * v;
    }
    scores[t] = sum;                                   // scratch use of scores[]
    scores[128 + t] = sumsq;
  }
  __syncthreads();
  if (t < 64) {
    float sum = scores[t] + scores[t + 64];
    float sq  = scores[128 + t] + scores[192 + t];
    float mu  = sum * (1.0f / CC);
    float var = sq * (1.0f / CC) - mu * mu;
    stat[t * 2]     = mu;
    stat[t * 2 + 1] = rsqrtf(var + 1e-5f);
  }
  __syncthreads();

  // ---- Phase 2: normalize -> bf16 tile ------------------------------------
  {
    const float mu = stat[srow * 2], rstd = stat[srow * 2 + 1];
    for (int c = cpar; c < CC; c += 2) {
      float v = regA[(size_t)srow * RA_STRIDE + c];
      xbf[(size_t)srow * XB_STRIDE + c] = f2bf((v - mu) * rstd * lnw[c] + lnb[c]);
    }
  }
  __syncthreads();

  // ---- Phase 3: K projection (WMMA) into regA -----------------------------
  project_wmma(xbf, bfWk, bk, regA, lane, wave);
  __syncthreads();

  // ---- Phase 4: scores[s][h] = scale * q_h . K[s, h*32 .. ] ---------------
  for (int i = t; i < SC * 8; i += 128) {
    const int sr = i >> 3, h = i & 7;
    const float* kr = regA + (size_t)sr * RA_STRIDE + h * 32;
    const float* qh = qsm + h * 32;
    float sc = 0.f;
#pragma unroll 8
    for (int d = 0; d < 32; ++d) sc += qh[d] * kr[d];
    scores[sr * 8 + h] = sc * 0.17677669529663687f;    // 1/sqrt(32)
  }
  __syncthreads();

  // ---- Phase 5: V projection (WMMA) into regA (K fully consumed) ----------
  project_wmma(xbf, bfWv, bv, regA, lane, wave);
  __syncthreads();

  // ---- Phase 6: chunk-local softmax partials ------------------------------
  {
    const int h  = t >> 4;                             // 0..7
    const int d0 = (t & 15) * 2;
    float m = -1e30f;
    for (int sr = 0; sr < SC; ++sr) m = fmaxf(m, scores[sr * 8 + h]);
    float l = 0.f, a0 = 0.f, a1 = 0.f;
    for (int sr = 0; sr < SC; ++sr) {
      float p = __expf(scores[sr * 8 + h] - m);
      l  += p;
      a0 += p * regA[(size_t)sr * RA_STRIDE + h * 32 + d0];
      a1 += p * regA[(size_t)sr * RA_STRIDE + h * 32 + d0 + 1];
    }
    float* pw = part_ws + ((size_t)bn * NCHUNK + chunk) * 272;
    if ((t & 15) == 0) { pw[h] = m; pw[8 + h] = l; }
    pw[16 + h * 32 + d0]     = a0;
    pw[16 + h * 32 + d0 + 1] = a1;
  }
}

// ---------------------------------------------------------------------------
// Kernel 3: merge partials -> ctx; attended = ctx@Wo.T+bo; updated; MLP; out.
// grid 64 (bn), block 256.
// ---------------------------------------------------------------------------
__global__ void epilogue_kernel(const float* __restrict__ mask_tokens,
                                const float* __restrict__ Wo,
                                const float* __restrict__ bo,
                                const float* __restrict__ W1,
                                const float* __restrict__ b1,
                                const float* __restrict__ W2,
                                const float* __restrict__ b2,
                                const float* __restrict__ part_ws,
                                float* __restrict__ out) {
  const int bn = blockIdx.x;
  const int t  = threadIdx.x;                          // 0..255
  __shared__ float ctx[CC];
  __shared__ float upd[CC];
  __shared__ float hid[2 * CC];

  // merge 64 chunks (log-sum-exp combine)
  {
    const int h = t >> 5, d = t & 31;
    const float* pw = part_ws + (size_t)bn * NCHUNK * 272;
    float m = -1e30f;
    for (int c = 0; c < NCHUNK; ++c) m = fmaxf(m, pw[(size_t)c * 272 + h]);
    float l = 0.f, acc = 0.f;
    for (int c = 0; c < NCHUNK; ++c) {
      const float* r = pw + (size_t)c * 272;
      float w = __expf(r[h] - m);
      l   += w * r[8 + h];
      acc += w * r[16 + h * 32 + d];
    }
    ctx[h * 32 + d] = acc / l;
  }
  __syncthreads();

  // attended + residual
  {
    float a = bo[t];
    const float* wr = Wo + (size_t)t * CC;
    for (int ci = 0; ci < CC; ++ci) a += ctx[ci] * wr[ci];
    upd[t] = mask_tokens[bn * CC + t] + a;
  }
  __syncthreads();

  // MLP hidden (2C = 512)
  for (int j = t; j < 2 * CC; j += 256) {
    float hsum = b1[j];
    const float* w1r = W1 + (size_t)j * CC;
    for (int ci = 0; ci < CC; ++ci) hsum += upd[ci] * w1r[ci];
    hid[j] = fmaxf(hsum, 0.f);
  }
  __syncthreads();

  {
    float o = b2[t];
    const float* w2r = W2 + (size_t)t * 2 * CC;
    for (int j = 0; j < 2 * CC; ++j) o += hid[j] * w2r[j];
    out[bn * CC + t] = upd[t] + o;
  }
}

// ---------------------------------------------------------------------------
extern "C" void kernel_launch(void* const* d_in, const int* in_sizes, int n_in,
                              void* d_out, int out_size, void* d_ws, size_t ws_size,
                              hipStream_t stream) {
  const float* mask_tokens = (const float*)d_in[0];
  const float* prompt      = (const float*)d_in[1];
  const float* ln_t_w = (const float*)d_in[2];
  const float* ln_t_b = (const float*)d_in[3];
  const float* ln_p_w = (const float*)d_in[4];
  const float* ln_p_b = (const float*)d_in[5];
  const float* Wq = (const float*)d_in[6];  const float* bq = (const float*)d_in[7];
  const float* Wk = (const float*)d_in[8];  const float* bk = (const float*)d_in[9];
  const float* Wv = (const float*)d_in[10]; const float* bv = (const float*)d_in[11];
  const float* Wo = (const float*)d_in[12]; const float* bo = (const float*)d_in[13];
  const float* W1 = (const float*)d_in[14]; const float* b1 = (const float*)d_in[15];
  const float* W2 = (const float*)d_in[16]; const float* b2 = (const float*)d_in[17];
  float* out = (float*)d_out;

  // workspace layout (floats): q[16384] | partials[64*64*272] | bf16 W [256KB]
  float* ws      = (float*)d_ws;
  float* q_ws    = ws;
  float* part_ws = ws + 16384;
  __bf16* bfW    = (__bf16*)(ws + 16384 + 64 * NCHUNK * 272);
  __bf16* bfWk   = bfW;
  __bf16* bfWv   = bfW + 65536;

  prep_weights_kernel<<<512, 256, 0, stream>>>(Wk, Wv, bfW);
  stage_q_kernel<<<64, 256, 0, stream>>>(mask_tokens, ln_t_w, ln_t_b, Wq, bq, q_ws);
  attn_partial_kernel<<<dim3(NCHUNK, 64), 128, SMEM_BYTES, stream>>>(
      prompt, ln_p_w, ln_p_b, bfWk, bk, bfWv, bv, q_ws, part_ws);
  epilogue_kernel<<<64, 256, 0, stream>>>(mask_tokens, Wo, bo, W1, b1, W2, b2,
                                          part_ws, out);
}
